// SequentialLoraA_59459527246470
// MI455X (gfx1250) — matrix operand order repaired and checked by reading, not compiled, not run on hardware
//
#include <hip/hip_runtime.h>
#include <hip/hip_bf16.h>

typedef __attribute__((ext_vector_type(16))) _Float16 v16h;
typedef __attribute__((ext_vector_type(8)))  float    v8f;
typedef __attribute__((ext_vector_type(4)))  unsigned int v4u;

#define DDIM 4096
#define RL 64
#define RS 16
#define B_LARGE 256
#define B_SMALL 256

union V16HCast { v4u u[2]; v16h h; };

// ---------------------------------------------------------------------------
// Large: one block per batch item. 4 waves, wave w owns N-tile w (16 cols).
// K chunked by 64; A-chunk (64x64 f16 = 8KB) staged transposed into LDS.
// ---------------------------------------------------------------------------
__global__ __launch_bounds__(128) void lora_large_kernel(
    const _Float16* __restrict__ x,     // (512, 4096) f16, use rows [0,256)
    const int*      __restrict__ wids,  // (256,)
    const _Float16* __restrict__ A,     // (128, 4096, 64) f16
    float*          __restrict__ out)   // (256, 64) f32
{
  constexpr int STRIDE = 72; // halves per shT row: 64 + 8 pad (bank-spread, 16B-aligned)
  __shared__ _Float16 xs[DDIM];
  __shared__ _Float16 shT[64 * STRIDE];

  const int b    = blockIdx.x;
  const int t    = threadIdx.x;   // 0..127
  const int lane = t & 31;
  const int wave = t >> 5;        // 0..3 == N-tile

  const _Float16* xg = x + (size_t)b * DDIM;
  const _Float16* Ab = A + (size_t)wids[b] * DDIM * RL;

  // stage x: 4096 halves = 512 x b128, coalesced
  {
    const v4u* src = (const v4u*)xg;
    v4u*       dst = (v4u*)xs;
#pragma unroll
    for (int i = 0; i < 4; ++i) dst[t + 128 * i] = src[t + 128 * i];
  }

  v8f acc = {0.f, 0.f, 0.f, 0.f, 0.f, 0.f, 0.f, 0.f};
  const int cSub  = lane & 15;     // N within tile / column of B
  const int half  = lane >> 4;     // which K-half this lane carries
  const int koffX = half * 16;     // x offset for A-operand (M=0 lanes)
  const int koffB = half * 16;     // K offset for B-operand

  for (int k0 = 0; k0 < DDIM; k0 += 64) {
    __syncthreads();
    // stage 64(K) x 64(R) tile transposed: 2 micro-tiles (2K x 8R) per thread
#pragma unroll
    for (int m2 = 0; m2 < 2; ++m2) {
      const int m  = t + 128 * m2;       // 0..255
      const int kk = (m >> 3) << 1;      // 0,2,..,62
      const int rr = (m & 7) << 3;       // 0,8,..,56
      const unsigned int* r0 = (const unsigned int*)(Ab + (size_t)(k0 + kk)     * RL + rr);
      const unsigned int* r1 = (const unsigned int*)(Ab + (size_t)(k0 + kk + 1) * RL + rr);
      if (k0 + 64 < DDIM)  // keep HBM stream ahead: global_prefetch_b8
        __builtin_prefetch((const void*)(Ab + (size_t)(k0 + 64 + kk) * RL + rr), 0, 1);
#pragma unroll
      for (int j2 = 0; j2 < 4; ++j2) {
        const unsigned int ar = r0[j2];   // A[kk ][rr+2j2], A[kk ][rr+2j2+1]
        const unsigned int br = r1[j2];   // A[kk+1][rr+2j2], A[kk+1][rr+2j2+1]
        const unsigned int lo = (ar & 0xffffu) | (br << 16);       // shT[r_lo][kk..kk+1]
        const unsigned int hi = (ar >> 16)     | (br & 0xffff0000u);
        const int r_lo = rr + 2 * j2;
        ((unsigned int*)shT)[(r_lo * STRIDE + kk) >> 1]       = lo;
        ((unsigned int*)shT)[((r_lo + 1) * STRIDE + kk) >> 1] = hi;
      }
    }
    __syncthreads();

    // two WMMAs per chunk per wave: K sub-steps 0 and 32
#pragma unroll
    for (int kkw = 0; kkw < 64; kkw += 32) {
      // A-operand: x in matrix row M=0 (lanes 0 and 16 per 16-bit A layout)
      V16HCast ax;
      const v4u z = {0u, 0u, 0u, 0u};
      ax.u[0] = z; ax.u[1] = z;
      if (cSub == 0) {
        const _Float16* xp = xs + k0 + kkw + koffX;
        ax.u[0] = *(const v4u*)xp;          // K = koffX..koffX+7
        ax.u[1] = *(const v4u*)(xp + 8);    // K = koffX+8..koffX+15
      }
      // B-operand: lane holds column cSub, K = kkw+koffB .. +15 (contiguous in shT)
      V16HCast bx;
      const _Float16* bp = shT + (wave * 16 + cSub) * STRIDE + kkw + koffB;
      bx.u[0] = *(const v4u*)bp;
      bx.u[1] = *(const v4u*)(bp + 8);
      acc = __builtin_amdgcn_wmma_f32_16x16x32_f16(false, ax.h, false, bx.h,
                                                   (short)0, acc, false, false);
    }
  }
  // D row M=0 lives in acc[0], lanes 0..15 (N = lane)
  if (lane < 16) out[(size_t)b * RL + wave * 16 + lane] = acc[0];
}

// ---------------------------------------------------------------------------
// Small: R=16 -> single N-tile; 4 waves split K, reduced via LDS.
// K chunked by 128; A-chunk (128x16 f16 = 4KB) staged transposed.
// ---------------------------------------------------------------------------
__global__ __launch_bounds__(128) void lora_small_kernel(
    const _Float16* __restrict__ x,     // (512, 4096), use rows [256,512)
    const int*      __restrict__ wids,  // (256,)
    const _Float16* __restrict__ A,     // (128, 4096, 16) f16
    float*          __restrict__ out)   // (256, 16) f32
{
  constexpr int STRIDE = 136; // halves per shT row: 128 + 8 pad
  __shared__ _Float16 xs[DDIM];
  __shared__ _Float16 shT[16 * STRIDE];
  __shared__ float    red[4][16];

  const int b    = blockIdx.x;
  const int t    = threadIdx.x;
  const int lane = t & 31;
  const int wave = t >> 5;

  const _Float16* xg = x + (size_t)(B_LARGE + b) * DDIM;
  const _Float16* Ab = A + (size_t)wids[b] * DDIM * RS;

  {
    const v4u* src = (const v4u*)xg;
    v4u*       dst = (v4u*)xs;
#pragma unroll
    for (int i = 0; i < 4; ++i) dst[t + 128 * i] = src[t + 128 * i];
  }

  v8f acc = {0.f, 0.f, 0.f, 0.f, 0.f, 0.f, 0.f, 0.f};
  const int cSub  = lane & 15;
  const int half  = lane >> 4;
  const int koffX = half * 16;
  const int koffB = half * 16;

  for (int k0 = 0; k0 < DDIM; k0 += 128) {
    __syncthreads();
    // stage 128(K) x 16(R) tile transposed: 1 micro-tile (2K x 8R) per thread
    {
      const int kk = (t >> 1) << 1;   // 0,2,..,126
      const int rr = (t & 1) << 3;    // 0 or 8
      const unsigned int* r0 = (const unsigned int*)(Ab + (size_t)(k0 + kk)     * RS + rr);
      const unsigned int* r1 = (const unsigned int*)(Ab + (size_t)(k0 + kk + 1) * RS + rr);
      if (k0 + 128 < DDIM)
        __builtin_prefetch((const void*)(Ab + (size_t)(k0 + 128 + kk) * RS + rr), 0, 1);
#pragma unroll
      for (int j2 = 0; j2 < 4; ++j2) {
        const unsigned int ar = r0[j2];
        const unsigned int br = r1[j2];
        const unsigned int lo = (ar & 0xffffu) | (br << 16);
        const unsigned int hi = (ar >> 16)     | (br & 0xffff0000u);
        const int r_lo = rr + 2 * j2;
        ((unsigned int*)shT)[(r_lo * STRIDE + kk) >> 1]       = lo;
        ((unsigned int*)shT)[((r_lo + 1) * STRIDE + kk) >> 1] = hi;
      }
    }
    __syncthreads();

    // each wave does one WMMA on its 32-wide K slice of the chunk
    {
      const int kkw = wave * 32;
      V16HCast ax;
      const v4u z = {0u, 0u, 0u, 0u};
      ax.u[0] = z; ax.u[1] = z;
      if (cSub == 0) {
        const _Float16* xp = xs + k0 + kkw + koffX;
        ax.u[0] = *(const v4u*)xp;
        ax.u[1] = *(const v4u*)(xp + 8);
      }
      V16HCast bx;
      const _Float16* bp = shT + cSub * STRIDE + kkw + koffB;
      bx.u[0] = *(const v4u*)bp;
      bx.u[1] = *(const v4u*)(bp + 8);
      acc = __builtin_amdgcn_wmma_f32_16x16x32_f16(false, ax.h, false, bx.h,
                                                   (short)0, acc, false, false);
    }
  }

  if (lane < 16) red[wave][lane] = acc[0];
  __syncthreads();
  if (t < 16)
    out[(size_t)b * RS + t] = red[0][t] + red[1][t] + red[2][t] + red[3][t];
}

extern "C" void kernel_launch(void* const* d_in, const int* in_sizes, int n_in,
                              void* d_out, int out_size, void* d_ws, size_t ws_size,
                              hipStream_t stream) {
  const _Float16* x  = (const _Float16*)d_in[0];
  const int*      wl = (const int*)d_in[1];
  const int*      wsm= (const int*)d_in[2];
  const _Float16* Al = (const _Float16*)d_in[3];
  const _Float16* As = (const _Float16*)d_in[4];
  float* out = (float*)d_out;

  lora_large_kernel<<<B_LARGE, 128, 0, stream>>>(x, wl, Al, out);
  lora_small_kernel<<<B_SMALL, 128, 0, stream>>>(x, wsm, As, out + (size_t)B_LARGE * RL);
}
